// VBrRenderer_36696200577473
// MI455X (gfx1250) — matrix operand order repaired and checked by reading, not compiled, not run on hardware
//
#include <hip/hip_runtime.h>
#include <hip/hip_bf16.h>
#include <math.h>

#define RB 8192      // rays
#define RN 256       // samples per ray
#define KPAD 96      // 82 padded to 3x32
#define HID 128

typedef __attribute__((ext_vector_type(16))) _Float16 v16h;
typedef __attribute__((ext_vector_type(8)))  _Float16 v8h;
typedef __attribute__((ext_vector_type(2)))  _Float16 v2h;
typedef __attribute__((ext_vector_type(8)))  float    v8f;

__device__ inline v8f vzero8f() {
  v8f z;
#pragma unroll
  for (int i = 0; i < 8; ++i) z[i] = 0.f;
  return z;
}
__device__ inline v16h vzero16h() {
  v16h z;
#pragma unroll
  for (int i = 0; i < 16; ++i) z[i] = (_Float16)0.f;
  return z;
}
// join two 8-half LDS loads into one 16-half WMMA operand
__device__ inline v16h lds16(const _Float16* p0, const _Float16* p1) {
  v8h lo = *(const v8h*)p0;
  v8h hi = *(const v8h*)p1;
  return __builtin_shufflevector(lo, hi, 0,1,2,3,4,5,6,7,8,9,10,11,12,13,14,15);
}

// ---------------- kernel 1: per-ray positional encoding table ----------------
__global__ void pe_kernel(const float* __restrict__ rays_d, _Float16* __restrict__ pe) {
  int r = blockIdx.x * blockDim.x + threadIdx.x;
  if (r >= RB) return;
  float d0 = rays_d[r*3+0], d1 = rays_d[r*3+1], d2 = rays_d[r*3+2];
  _Float16* o = pe + (size_t)r * 64;
  float b = 1.f;
#pragma unroll
  for (int f = 0; f < 9; ++f) {
    float x0 = d0*b, x1 = d1*b, x2 = d2*b;
    o[f*6+0] = (_Float16)sinf(x0);
    o[f*6+1] = (_Float16)sinf(x1);
    o[f*6+2] = (_Float16)sinf(x2);
    o[f*6+3] = (_Float16)cosf(x0);
    o[f*6+4] = (_Float16)cosf(x1);
    o[f*6+5] = (_Float16)cosf(x2);
    b *= 2.f;
  }
  for (int i = 54; i < 64; ++i) o[i] = (_Float16)0.f;
}

// ---------------- kernel 2: per-ray masked-sample counts ----------------
__global__ void count_kernel(const unsigned char* __restrict__ mask,
                             unsigned int* __restrict__ cnt) {
  int r = blockIdx.x * blockDim.x + threadIdx.x;
  if (r >= RB) return;
  const unsigned int* mw = (const unsigned int*)(mask + (size_t)r * RN);
  unsigned s = 0;
#pragma unroll
  for (int i = 0; i < RN/4; ++i) s += __popc(mw[i]);   // bytes are 0/1
  cnt[r] = s;
}

// ---------------- kernel 3: exclusive scan over 8192 counts (1 block) -------
__global__ void __launch_bounds__(1024) scan_kernel(const unsigned int* __restrict__ cnt,
                                                    unsigned int* __restrict__ start) {
  __shared__ unsigned int part[1024];
  int t = threadIdx.x;
  unsigned c[8];
  unsigned run = 0;
#pragma unroll
  for (int i = 0; i < 8; ++i) {
    unsigned v = cnt[t*8 + i];
    c[i] = run;           // local exclusive
    run += v;
  }
  part[t] = run;
  __syncthreads();
  for (int off = 1; off < 1024; off <<= 1) {
    unsigned v = (t >= off) ? part[t - off] : 0u;
    __syncthreads();
    part[t] += v;
    __syncthreads();
  }
  unsigned base = part[t] - run;   // exclusive prefix for this thread's chunk
#pragma unroll
  for (int i = 0; i < 8; ++i) start[t*8 + i] = base + c[i];
}

// ---------------- kernel 4: point -> ray id (compaction order) ----------------
__global__ void __launch_bounds__(128) fill_kernel(const unsigned char* __restrict__ mask,
                                                   const unsigned int* __restrict__ ray_start,
                                                   unsigned int* __restrict__ point_ray) {
  int wave = threadIdx.x >> 5, lane = threadIdx.x & 31;
  int r = blockIdx.x * 4 + wave;
  if (r >= RB) return;
  unsigned base = ray_start[r];
#pragma unroll
  for (int ch = 0; ch < 8; ++ch) {
    int j = ch*32 + lane;
    int m = mask[(size_t)r*RN + j] ? 1 : 0;
    unsigned bal = (unsigned)__ballot(m);
    if (m) point_ray[base + __popc(bal & ((1u << lane) - 1u))] = (unsigned)r;
    base += __popc(bal);
  }
}

// ---------------- kernel 5: WMMA MLP over 16-point tiles ----------------
__global__ void __launch_bounds__(128, 1) mlp_kernel(
    const float* __restrict__ dataP, const _Float16* __restrict__ pe,
    const unsigned int* __restrict__ point_ray,
    const float* __restrict__ W1g, const float* __restrict__ b1g,
    const float* __restrict__ W2g, const float* __restrict__ b2g,
    float* __restrict__ densc, float* __restrict__ colc, int P)
{
  __shared__ _Float16 sW1[HID * KPAD];   // 24 KB staging, row-major padded
  __shared__ _Float16 sW2[4 * HID];      // 1 KB staging
  __shared__ float    sB1[HID];
  __shared__ _Float16 sA[4][16 * KPAD];  // per-wave A tile
  __shared__ _Float16 sH[4][16 * HID];   // per-wave h tile

  const int tid = threadIdx.x;
  {  // cooperative weight staging (128 threads, one W1 row each)
    int j = tid;
    for (int k = 0; k < KPAD; ++k)
      sW1[j*KPAD + k] = (k < 82) ? (_Float16)W1g[j*82 + k] : (_Float16)0.f;
    if (j < 4)
      for (int k = 0; k < HID; ++k) sW2[j*HID + k] = (_Float16)W2g[j*HID + k];
    sB1[j] = b1g[j];
  }
  __syncthreads();

  const int wave = tid >> 5;
  const int lane = tid & 31;
  const int n    = lane & 15;
  const int hi   = lane >> 4;
  const int gw     = blockIdx.x * 4 + wave;
  const int nWaves = gridDim.x * 4;
  const int nTiles = (P + 15) / 16;
  _Float16* As = sA[wave];
  _Float16* Hs = sH[wave];
  const float bb2 = (n < 4) ? b2g[n] : 0.f;

  // ---- hoist ALL loop-invariant B operands + biases into registers ----
  v16h Bw1[24];                         // layer-1 W1 operands: [c*8 + nt]
#pragma unroll
  for (int c = 0; c < 3; ++c)
#pragma unroll
    for (int nt = 0; nt < 8; ++nt) {
      const _Float16* pb = &sW1[(nt*16 + n)*KPAD + c*32 + hi*16];
      Bw1[c*8 + nt] = lds16(pb, pb + 8);  // B layout: 16 contiguous K per lane
    }
  v16h Bw2[4];                          // layer-2 W2 operands (zero lanes n>=4)
#pragma unroll
  for (int c = 0; c < 4; ++c) {
    if (n < 4) {
      const _Float16* pb = &sW2[n*HID + c*32 + hi*16];
      Bw2[c] = lds16(pb, pb + 8);
    } else {
      Bw2[c] = vzero16h();
    }
  }
  float biasr[8];
#pragma unroll
  for (int nt = 0; nt < 8; ++nt) biasr[nt] = sB1[nt*16 + n];

  const _Float16 hz = (_Float16)0.f;

  for (int t = gw; t < nTiles; t += nWaves) {
    // ---- build A tile (2 lanes per point; one uniform branch on hi) ----
    {
      int p0 = t*16 + n;
      bool valid = p0 < P;
      int p = valid ? p0 : (P - 1);      // clamp: loads always in-bounds
      unsigned r = point_ray[p];
      const float2*   dr2 = (const float2*)(dataP + (size_t)p * 28);
      const _Float16* per = pe + (size_t)r * 64;
      _Float16* arow = &As[n*KPAD];
      if (hi == 0) {
#pragma unroll
        for (int q = 0; q < 14; ++q) {       // k = 0..27 : data features
          float2 d2 = dr2[q];
          v2h pr;
          pr[0] = valid ? (_Float16)d2.x : hz;
          pr[1] = valid ? (_Float16)d2.y : hz;
          *(v2h*)(arow + 2*q) = pr;
        }
#pragma unroll
        for (int q = 14; q < 24; ++q) {      // k = 28..47 : pe[0..19]
          v2h pr = *(const v2h*)(per + (2*q - 28));
          pr[0] = valid ? pr[0] : hz;
          pr[1] = valid ? pr[1] : hz;
          *(v2h*)(arow + 2*q) = pr;
        }
      } else {
#pragma unroll
        for (int q = 24; q < 41; ++q) {      // k = 48..80(,81) : pe[20..53]
          v2h pr = *(const v2h*)(per + (2*q - 28));
          pr[0] = valid ? pr[0] : hz;
          pr[1] = valid ? pr[1] : hz;
          *(v2h*)(arow + 2*q) = pr;
        }
#pragma unroll
        for (int q = 41; q < 48; ++q) {      // k = 82..95 : zero pad
          v2h pr; pr[0] = hz; pr[1] = hz;
          *(v2h*)(arow + 2*q) = pr;
        }
      }
    }
    asm volatile("" ::: "memory");   // same-wave DS store->load ordering

    // ---- layer 1: h(16x128) = A(16x96) @ W1^T ; B already in registers ----
    v8f acc[8];
#pragma unroll
    for (int i = 0; i < 8; ++i) acc[i] = vzero8f();
#pragma unroll
    for (int c = 0; c < 3; ++c) {
      const _Float16* pa = &As[n*KPAD + c*32 + hi*16];
      v16h a = lds16(pa, pa + 8);   // A layout: 16 contiguous K per lane
#pragma unroll
      for (int nt = 0; nt < 8; ++nt) {
        acc[nt] = __builtin_amdgcn_wmma_f32_16x16x32_f16(
            false, a, false, Bw1[c*8 + nt], (short)0, acc[nt], false, false);
      }
    }
    // bias + relu + f16, store h tile row-major to LDS
#pragma unroll
    for (int nt = 0; nt < 8; ++nt) {
      float bias = biasr[nt];
#pragma unroll
      for (int v = 0; v < 8; ++v) {
        float h = acc[nt][v] + bias;
        h = h > 0.f ? h : 0.f;
        Hs[(v + 8*hi)*HID + nt*16 + n] = (_Float16)h;
      }
    }
    asm volatile("" ::: "memory");

    // ---- layer 2: preact(16x16, cols 0..3 live) = h(16x128) @ W2pad^T ----
    v8f acc2 = vzero8f();
#pragma unroll
    for (int c = 0; c < 4; ++c) {
      const _Float16* pa = &Hs[n*HID + c*32 + hi*16];
      v16h a2 = lds16(pa, pa + 8);
      acc2 = __builtin_amdgcn_wmma_f32_16x16x32_f16(
          false, a2, false, Bw2[c], (short)0, acc2, false, false);
    }
    // epilogue: lane column n<4 owns output channel n, rows v + 8*hi
    if (n < 4) {
#pragma unroll
      for (int v = 0; v < 8; ++v) {
        int p2 = t*16 + v + 8*hi;
        if (p2 < P) {
          float val = acc2[v] + bb2;
          if (n == 0) densc[p2] = val > 0.f ? val : 0.f;
          else        colc[(size_t)p2*3 + (n-1)] = val;
        }
      }
    }
  }
}

// ---------------- kernel 6: per-ray compositing (wave32 per ray) -------------
__global__ void __launch_bounds__(128) composite_kernel(
    const float* __restrict__ rays_d, const float* __restrict__ intrs,
    const unsigned char* __restrict__ mask, const unsigned int* __restrict__ ray_start,
    const float* __restrict__ densc, const float* __restrict__ colc,
    float* __restrict__ out_comp, float* __restrict__ out_alpha)
{
  int wave = threadIdx.x >> 5, lane = threadIdx.x & 31;
  int r = blockIdx.x * 4 + wave;
  if (r >= RB) return;
  float dx = rays_d[r*3+0], dy = rays_d[r*3+1], dz = rays_d[r*3+2];
  float dn = sqrtf(dx*dx + dy*dy + dz*dz);
  unsigned pbase = ray_start[r];
  float T = 1.f, asum = 0.f, c0 = 0.f, c1 = 0.f, c2 = 0.f;
#pragma unroll 1
  for (int ch = 0; ch < 8; ++ch) {
    int j = ch*32 + lane;
    float t0 = intrs[(size_t)r*(RN+1) + j];
    float t1 = intrs[(size_t)r*(RN+1) + j + 1];
    float dist = (t1 - t0) * dn;
    int m = mask[(size_t)r*RN + j] ? 1 : 0;
    unsigned bal = (unsigned)__ballot(m);
    unsigned rank = __popc(bal & ((1u << lane) - 1u));
    float d = 0.f;
    if (m) d = densc[pbase + rank];
    float alpha = 1.f - __expf(-d * dist);
    float pf = 1.f - alpha + 1e-10f;
    float x = pf;                                // inclusive wave prefix product
#pragma unroll
    for (int off = 1; off < 32; off <<= 1) {
      float s = __shfl_up(x, off, 32);
      if (lane >= off) x *= s;
    }
    float excl = __shfl_up(x, 1, 32);
    if (lane == 0) excl = 1.f;
    float absl = alpha * (T * excl);
    asum += absl;
    if (m && absl != 0.f) {
      size_t cb = (size_t)(pbase + rank) * 3;
      c0 += absl * (1.f / (1.f + __expf(-colc[cb+0])));
      c1 += absl * (1.f / (1.f + __expf(-colc[cb+1])));
      c2 += absl * (1.f / (1.f + __expf(-colc[cb+2])));
    }
    out_alpha[(size_t)r*RN + j] = alpha;
    T *= __shfl(x, 31, 32);
    pbase += (unsigned)__popc(bal);
  }
#pragma unroll
  for (int off = 16; off; off >>= 1) {
    c0   += __shfl_xor(c0, off, 32);
    c1   += __shfl_xor(c1, off, 32);
    c2   += __shfl_xor(c2, off, 32);
    asum += __shfl_xor(asum, off, 32);
  }
  if (lane == 0) {
    float bg = 1.f - asum;
    out_comp[r*3+0] = c0 + bg;
    out_comp[r*3+1] = c1 + bg;
    out_comp[r*3+2] = c2 + bg;
  }
}

extern "C" void kernel_launch(void* const* d_in, const int* in_sizes, int n_in,
                              void* d_out, int out_size, void* d_ws, size_t ws_size,
                              hipStream_t stream) {
  (void)n_in; (void)out_size; (void)ws_size;
  const float*         rays_d = (const float*)d_in[0];
  const float*         intrs  = (const float*)d_in[1];
  const unsigned char* mask   = (const unsigned char*)d_in[2];
  const float*         dataP  = (const float*)d_in[3];
  const float*         W1     = (const float*)d_in[4];
  const float*         b1     = (const float*)d_in[5];
  const float*         W2     = (const float*)d_in[6];
  const float*         b2     = (const float*)d_in[7];
  const int P = in_sizes[3] / 28;

  // workspace carve (256B aligned)
  char* ws = (char*)d_ws;
  size_t off = 0;
  auto carve = [&](size_t bytes) {
    void* p = ws + off;
    off = (off + bytes + 255) & ~(size_t)255;
    return p;
  };
  unsigned int* ray_cnt   = (unsigned int*)carve((size_t)RB * 4);
  unsigned int* ray_start = (unsigned int*)carve((size_t)RB * 4);
  unsigned int* point_ray = (unsigned int*)carve((size_t)P * 4);
  _Float16*     pe        = (_Float16*)carve((size_t)RB * 64 * 2);
  float*        densc     = (float*)carve((size_t)P * 4);
  float*        colc      = (float*)carve((size_t)P * 12);

  float* out_comp  = (float*)d_out;
  float* out_alpha = out_comp + (size_t)RB * 3;

  pe_kernel<<<RB/256, 256, 0, stream>>>(rays_d, pe);
  count_kernel<<<RB/256, 256, 0, stream>>>(mask, ray_cnt);
  scan_kernel<<<1, 1024, 0, stream>>>(ray_cnt, ray_start);
  fill_kernel<<<RB/4, 128, 0, stream>>>(mask, ray_start, point_ray);
  mlp_kernel<<<512, 128, 0, stream>>>(dataP, pe, point_ray, W1, b1, W2, b2,
                                      densc, colc, P);
  composite_kernel<<<RB/4, 128, 0, stream>>>(rays_d, intrs, mask, ray_start,
                                             densc, colc, out_comp, out_alpha);
}
